// SoftmaxQuantize_4174708212513
// MI455X (gfx1250) — matrix operand order repaired and checked by reading, not compiled, not run on hardware
//
#include <hip/hip_runtime.h>

// ---------------------------------------------------------------------------
// Fused LayerNorm -> soft-quantize attention (Q = LN(x), K = V = embd)
// Flash-attention style streaming over N=8192 with online softmax + argmax.
// GEMMs run on v_wmma_f32_16x16x32_bf16 (f32 accumulate).
// Logits are produced directly in exp2 domain (scale*log2e folded into X).
// ---------------------------------------------------------------------------

typedef __attribute__((ext_vector_type(16))) __bf16 v16bf;
typedef __attribute__((ext_vector_type(8)))  __bf16 v8bf;
typedef __attribute__((ext_vector_type(8)))  float  v8f;

#define NEMB 8192
#define DIM  256
#define ROWS 16384   // B*T = 16*1024

// 1/sqrt(256) * log2(e): logits * this => softmax via exp2
#define KSCALE 0.09013878188659973f

__device__ __forceinline__ float fast_exp2(float x) {
#if __has_builtin(__builtin_amdgcn_exp2f)
    return __builtin_amdgcn_exp2f(x);
#else
    return exp2f(x);
#endif
}

__device__ __forceinline__ unsigned short f2bf_bits(float f) {
    __bf16 h = (__bf16)f;                  // native v_cvt (RNE)
    unsigned short s;
    __builtin_memcpy(&s, &h, 2);
    return s;
}

__device__ __forceinline__ uint4 bc_u4(v8bf v) { uint4 u; __builtin_memcpy(&u, &v, 16); return u; }
__device__ __forceinline__ v8bf  bc_v8(uint4 u) { v8bf v; __builtin_memcpy(&v, &u, 16); return v; }

// --------------------------- Prep 1: LayerNorm -> bf16 ---------------------
// One wave per row; lane handles 8 contiguous elements. Output pre-scaled by
// KSCALE so the S^T GEMM lands in exp2 domain.
__global__ __launch_bounds__(256) void ln_bf16_kernel(
    const float* __restrict__ x, const float* __restrict__ w,
    const float* __restrict__ b, unsigned short* __restrict__ Xbf)
{
    const int row  = blockIdx.x * 8 + (threadIdx.x >> 5);
    const int lane = threadIdx.x & 31;
    const float* xr = x + (size_t)row * DIM + lane * 8;
    float4 a0 = *(const float4*)(xr);
    float4 a1 = *(const float4*)(xr + 4);

    float s = a0.x + a0.y + a0.z + a0.w + a1.x + a1.y + a1.z + a1.w;
    float q = a0.x*a0.x + a0.y*a0.y + a0.z*a0.z + a0.w*a0.w
            + a1.x*a1.x + a1.y*a1.y + a1.z*a1.z + a1.w*a1.w;
    #pragma unroll
    for (int o = 16; o > 0; o >>= 1) {
        s += __shfl_xor(s, o, 32);
        q += __shfl_xor(q, o, 32);
    }
    const float mean = s * (1.0f / DIM);
    const float var  = q * (1.0f / DIM) - mean * mean;
    const float inv  = rsqrtf(var + 1e-5f);

    float4 w0 = *(const float4*)(w + lane * 8);
    float4 w1 = *(const float4*)(w + lane * 8 + 4);
    float4 b0 = *(const float4*)(b + lane * 8);
    float4 b1 = *(const float4*)(b + lane * 8 + 4);

    float y[8];
    y[0] = ((a0.x - mean) * inv * w0.x + b0.x) * KSCALE;
    y[1] = ((a0.y - mean) * inv * w0.y + b0.y) * KSCALE;
    y[2] = ((a0.z - mean) * inv * w0.z + b0.z) * KSCALE;
    y[3] = ((a0.w - mean) * inv * w0.w + b0.w) * KSCALE;
    y[4] = ((a1.x - mean) * inv * w1.x + b1.x) * KSCALE;
    y[5] = ((a1.y - mean) * inv * w1.y + b1.y) * KSCALE;
    y[6] = ((a1.z - mean) * inv * w1.z + b1.z) * KSCALE;
    y[7] = ((a1.w - mean) * inv * w1.w + b1.w) * KSCALE;

    uint4 pk;
    pk.x = (unsigned)f2bf_bits(y[0]) | ((unsigned)f2bf_bits(y[1]) << 16);
    pk.y = (unsigned)f2bf_bits(y[2]) | ((unsigned)f2bf_bits(y[3]) << 16);
    pk.z = (unsigned)f2bf_bits(y[4]) | ((unsigned)f2bf_bits(y[5]) << 16);
    pk.w = (unsigned)f2bf_bits(y[6]) | ((unsigned)f2bf_bits(y[7]) << 16);
    *(uint4*)(Xbf + (size_t)row * DIM + lane * 8) = pk;
}

// ------------------- Prep 2: embd -> bf16 (row-major + transposed) ---------
__global__ __launch_bounds__(256) void embd_cvt_kernel(
    const float* __restrict__ e,
    unsigned short* __restrict__ Ebf, unsigned short* __restrict__ ETbf)
{
    const int n = blockIdx.x;
    const int d = threadIdx.x;
    unsigned short s = f2bf_bits(e[(size_t)n * DIM + d]);
    Ebf[(size_t)n * DIM + d]   = s;
    ETbf[(size_t)d * NEMB + n] = s;
}

// --------------------------- Fused attention kernel ------------------------
// One wave per 16-row query tile. Stream N in super-chunks of 32.
//   S^T  = E_chunk(16x32 over d, A-layout) x X^T(32x16, B-layout)   [x16 WMMA]
//   O^T += E^T_tile(16x32 over n, A-layout) x P^T(32x16, B-layout)  [x16 WMMA]
// C-layout of S^T puts query row m in lane (lane%16) -> softmax is in-lane.
__global__ __launch_bounds__(128) void fused_attn_kernel(
    const unsigned short* __restrict__ Xbf,
    const unsigned short* __restrict__ Ebf,
    const unsigned short* __restrict__ ETbf,
    float* __restrict__ outq, float* __restrict__ outi)
{
    const int lane = threadIdx.x & 31;
    const int wave = threadIdx.x >> 5;
    const int tile = blockIdx.x * 4 + wave;
    const int lp   = lane & 15;   // query row within tile / B-matrix column
    const int half = lane >> 4;
    const int mrow = tile * 16 + lp;

    // Resident B-matrix operand: X^T. B layout: lane holds col=lp,
// j<8 -> K(d) = 32k + 8*half + j, j>=8 -> K(d) = 32k + 16 + 8*half + (j-8).
    v16bf Bx[8];
    {
        const unsigned short* xr = Xbf + (size_t)mrow * DIM;
        #pragma unroll
        for (int k = 0; k < 8; ++k) {
            v8bf lo = *(const v8bf*)(xr + 32 * k + 8 * half);
            v8bf hi = *(const v8bf*)(xr + 32 * k + 16 + 8 * half);
            #pragma unroll
            for (int i = 0; i < 8; ++i) { Bx[k][i] = lo[i]; Bx[k][8 + i] = hi[i]; }
        }
    }

    v8f acc[16];                               // O^T accumulators, 16 d-tiles
    #pragma unroll
    for (int t = 0; t < 16; ++t)
        #pragma unroll
        for (int v = 0; v < 8; ++v) acc[t][v] = 0.0f;

    float Mr = -3.0e38f, Lr = 0.0f;
    float bestv = -3.0e38f; int bestn = 0;

    for (int sc = 0; sc < NEMB / 32; ++sc) {
        const int n0 = sc * 32;

        // ---- S^T for 32 n-values (two 16-col chunks); already exp2-domain --
        float S0[8], S1[8];
        #pragma unroll
        for (int c = 0; c < 2; ++c) {
            v8f s = {0.f, 0.f, 0.f, 0.f, 0.f, 0.f, 0.f, 0.f};
            // A-layout: lane row = n-local = lp; halves j<8 -> d=32k+8*half+j,
            // j>=8 -> d=32k+16+8*half+(j-8).
            const unsigned short* er = Ebf + (size_t)(n0 + 16 * c + lp) * DIM;
            #pragma unroll
            for (int k = 0; k < 8; ++k) {
                v8bf lo = *(const v8bf*)(er + 32 * k + 8 * half);
                v8bf hi = *(const v8bf*)(er + 32 * k + 16 + 8 * half);
                v16bf a;
                #pragma unroll
                for (int i = 0; i < 8; ++i) { a[i] = lo[i]; a[8 + i] = hi[i]; }
                s = __builtin_amdgcn_wmma_f32_16x16x32_bf16(
                        false, a, false, Bx[k], (short)0, s, false, false);
            }
            #pragma unroll
            for (int v = 0; v < 8; ++v) {
                if (c == 0) S0[v] = s[v]; else S1[v] = s[v];
            }
        }

        // ---- per-lane chunk max (needed for softmax anyway) ----
        float m_own = fmaxf(S0[0], S1[0]);
        #pragma unroll
        for (int v = 1; v < 8; ++v) m_own = fmaxf(m_own, fmaxf(S0[v], S1[v]));

        // ---- argmax index scan only when some lane has a new record ----
        if (__any(m_own > bestv)) {
            #pragma unroll
            for (int v = 0; v < 8; ++v) {
                const int nA = n0 + v + 8 * half;
                const int nB = n0 + 16 + v + 8 * half;
                if (S0[v] > bestv) { bestv = S0[v]; bestn = nA; }
                if (S1[v] > bestv) { bestv = S1[v]; bestn = nB; }
            }
        }

        // ---- online softmax (rows live in lane pairs {lp, lp+16}) ----
        const float mc   = fmaxf(m_own, __shfl_xor(m_own, 16, 32));
        const float Mnew = fmaxf(Mr, mc);
        const float alpha = fast_exp2(Mr - Mnew);

        float P0[8], P1[8], lsum = 0.0f;
        #pragma unroll
        for (int v = 0; v < 8; ++v) {
            P0[v] = fast_exp2(S0[v] - Mnew);
            P1[v] = fast_exp2(S1[v] - Mnew);
            lsum += P0[v] + P1[v];
        }
        lsum += __shfl_xor(lsum, 16, 32);
        Lr = Lr * alpha + lsum;
        Mr = Mnew;

        // rescale O^T only when some row's max actually moved (alpha < 1)
        if (__any(alpha != 1.0f)) {
            #pragma unroll
            for (int t = 0; t < 16; ++t)
                #pragma unroll
                for (int v = 0; v < 8; ++v) acc[t][v] *= alpha;
        }

        // ---- pack P^T into B layout: lane col = lp, K(n) = n0 + 16*half + j.
        // Convert to bf16 first, then swap packed dwords across the lane pair:
        // half=0 needs partner's P0 for j=8..15; half=1 needs partner's P1
        // for j=0..7.
        v8bf own0, own1;
        #pragma unroll
        for (int v = 0; v < 8; ++v) { own0[v] = (__bf16)P0[v]; own1[v] = (__bf16)P1[v]; }
        uint4 u0 = bc_u4(own0), u1 = bc_u4(own1), q0, q1;
        q0.x = __shfl_xor(u0.x, 16, 32); q0.y = __shfl_xor(u0.y, 16, 32);
        q0.z = __shfl_xor(u0.z, 16, 32); q0.w = __shfl_xor(u0.w, 16, 32);
        q1.x = __shfl_xor(u1.x, 16, 32); q1.y = __shfl_xor(u1.y, 16, 32);
        q1.z = __shfl_xor(u1.z, 16, 32); q1.w = __shfl_xor(u1.w, 16, 32);
        uint4 plo, phi;
        plo.x = half ? q1.x : u0.x; plo.y = half ? q1.y : u0.y;
        plo.z = half ? q1.z : u0.z; plo.w = half ? q1.w : u0.w;
        phi.x = half ? u1.x : q0.x; phi.y = half ? u1.y : q0.y;
        phi.z = half ? u1.z : q0.z; phi.w = half ? u1.w : q0.w;
        v8bf pl = bc_v8(plo), ph = bc_v8(phi);
        v16bf Bp;
        #pragma unroll
        for (int i = 0; i < 8; ++i) { Bp[i] = pl[i]; Bp[8 + i] = ph[i]; }

        // ---- O^T += E^T_tile x P^T for all 16 d-tiles ----
        #pragma unroll
        for (int t = 0; t < 16; ++t) {
            const unsigned short* tr = ETbf + (size_t)(t * 16 + lp) * NEMB + n0;
            v8bf lo = *(const v8bf*)(tr + 16 * half);
            v8bf hi = *(const v8bf*)(tr + 16 * half + 8);
            v16bf a;
            #pragma unroll
            for (int i = 0; i < 8; ++i) { a[i] = lo[i]; a[8 + i] = hi[i]; }
            acc[t] = __builtin_amdgcn_wmma_f32_16x16x32_bf16(
                         false, a, false, Bp, (short)0, acc[t], false, false);
        }
    }

    // ---- epilogue: O = O^T / L, coalesced float4 stores ----
    const float inv = 1.0f / Lr;
    float* orow = outq + (size_t)mrow * DIM + 8 * half;  // d' = v + 8*half
    #pragma unroll
    for (int t = 0; t < 16; ++t) {
        float4 w0, w1;
        w0.x = acc[t][0] * inv; w0.y = acc[t][1] * inv;
        w0.z = acc[t][2] * inv; w0.w = acc[t][3] * inv;
        w1.x = acc[t][4] * inv; w1.y = acc[t][5] * inv;
        w1.z = acc[t][6] * inv; w1.w = acc[t][7] * inv;
        *(float4*)(orow + t * 16)     = w0;
        *(float4*)(orow + t * 16 + 4) = w1;
    }

    // ---- argmax: combine lane pair (prefer lower index on tie), write ----
    float ov = __shfl_xor(bestv, 16, 32);
    int   on = __shfl_xor(bestn, 16, 32);
    if (ov > bestv || (ov == bestv && on < bestn)) { bestv = ov; bestn = on; }
    if (half == 0) outi[tile * 16 + lp] = (float)bestn;
}

// ---------------------------------------------------------------------------
extern "C" void kernel_launch(void* const* d_in, const int* in_sizes, int n_in,
                              void* d_out, int out_size, void* d_ws, size_t ws_size,
                              hipStream_t stream) {
    const float* x    = (const float*)d_in[0];   // [16,1024,256]
    const float* lnw  = (const float*)d_in[1];   // [256]
    const float* lnb  = (const float*)d_in[2];   // [256]
    const float* embd = (const float*)d_in[3];   // [8192,256]
    float* out = (float*)d_out;                  // quantize [16384*256] ++ idxs [16384]

    unsigned short* Xbf  = (unsigned short*)d_ws;          // 8 MB
    unsigned short* Ebf  = Xbf + (size_t)ROWS * DIM;       // 4 MB
    unsigned short* ETbf = Ebf + (size_t)NEMB * DIM;       // 4 MB

    ln_bf16_kernel  <<<ROWS / 8, 256, 0, stream>>>(x, lnw, lnb, Xbf);
    embd_cvt_kernel <<<NEMB,     256, 0, stream>>>(embd, Ebf, ETbf);
    fused_attn_kernel<<<(ROWS / 16) / 4, 128, 0, stream>>>(
        Xbf, Ebf, ETbf, out, out + (size_t)ROWS * DIM);
}
